// SingleScaleNeUS_65541200937485
// MI455X (gfx1250) — matrix operand order repaired and checked by reading, not compiled, not run on hardware
//
#include <hip/hip_runtime.h>
#include <hip/hip_bf16.h>
#include <math.h>

typedef __attribute__((ext_vector_type(16))) _Float16 v16h;
typedef __attribute__((ext_vector_type(8)))  _Float16 v8h;
typedef __attribute__((ext_vector_type(8)))  float    v8f;
typedef _Float16 h16;

#define NVOX    262144      // 64^3
#define NVOXUP  2097152     // 128^3
#define NSAMP   447
#define NRAYS   4096
#define TOTS    (NRAYS*NSAMP)   // 1,830,912 = 14304*128 exactly
#define STEPVOX 0.015625f       // STEP * VOXEL = 0.5 * (2/64)

// ---------------------------------------------------------------- conv1 12->64
__global__ __launch_bounds__(256) void k_conv1(
    const float* __restrict__ grid, const float* __restrict__ w,
    const float* __restrict__ b, const float* __restrict__ g,
    const float* __restrict__ beta, const float* __restrict__ m,
    const float* __restrict__ v, float* __restrict__ x1)
{
    __shared__ float patch[648];                 // 12ch * 3 * 3 * 6 cols
    int p0 = blockIdx.x * 4;
    int d0 = p0 >> 12, h0 = (p0 >> 6) & 63, w0 = p0 & 63;
    for (int i = threadIdx.x; i < 648; i += 256) {
        int cc = i % 6; int r = i / 6; int yy = r % 3; r /= 3; int zz = r % 3; int ic = r / 3;
        int gz = d0 + zz - 1, gy = h0 + yy - 1, gx = w0 + cc - 1;
        float val = 0.f;
        if ((unsigned)gz < 64u && (unsigned)gy < 64u && (unsigned)gx < 64u)
            val = grid[ic * NVOX + (gz * 64 + gy) * 64 + gx];
        patch[i] = val;
    }
    __syncthreads();
    int oc = threadIdx.x & 63, pp = threadIdx.x >> 6;
    const float* wp = w + oc * 324;
    float acc = 0.f;
    for (int ic = 0; ic < 12; ++ic) {
        #pragma unroll
        for (int zz = 0; zz < 3; ++zz)
            #pragma unroll
            for (int yy = 0; yy < 3; ++yy)
                #pragma unroll
                for (int xx = 0; xx < 3; ++xx)
                    acc += patch[((ic * 3 + zz) * 3 + yy) * 6 + pp + xx] *
                           wp[ic * 27 + zz * 9 + yy * 3 + xx];
    }
    float s = g[oc] * rsqrtf(v[oc] + 1e-5f);
    float y = (acc + b[oc]) * s + (beta[oc] - m[oc] * s);
    x1[oc * NVOX + p0 + pp] = fmaxf(y, 0.f);
}

// ---------------------------------------------------------------- conv2 76->64
__global__ __launch_bounds__(256) void k_conv2(
    const float* __restrict__ grid, const float* __restrict__ x1,
    const float* __restrict__ w, const float* __restrict__ b,
    const float* __restrict__ g, const float* __restrict__ beta,
    const float* __restrict__ m, const float* __restrict__ v,
    float* __restrict__ x2)
{
    __shared__ float patch[4104];                // 76ch * 3 * 3 * 6 cols
    int p0 = blockIdx.x * 4;
    int d0 = p0 >> 12, h0 = (p0 >> 6) & 63, w0 = p0 & 63;
    for (int i = threadIdx.x; i < 4104; i += 256) {
        int cc = i % 6; int r = i / 6; int yy = r % 3; r /= 3; int zz = r % 3; int ic = r / 3;
        int gz = d0 + zz - 1, gy = h0 + yy - 1, gx = w0 + cc - 1;
        float val = 0.f;
        if ((unsigned)gz < 64u && (unsigned)gy < 64u && (unsigned)gx < 64u) {
            int idx = (gz * 64 + gy) * 64 + gx;
            val = (ic < 12) ? grid[ic * NVOX + idx] : x1[(ic - 12) * NVOX + idx];
        }
        patch[i] = val;
    }
    __syncthreads();
    int oc = threadIdx.x & 63, pp = threadIdx.x >> 6;
    const float* wp = w + oc * 2052;
    float acc = 0.f;
    for (int ic = 0; ic < 76; ++ic) {
        #pragma unroll
        for (int zz = 0; zz < 3; ++zz)
            #pragma unroll
            for (int yy = 0; yy < 3; ++yy)
                #pragma unroll
                for (int xx = 0; xx < 3; ++xx)
                    acc += patch[((ic * 3 + zz) * 3 + yy) * 6 + pp + xx] *
                           wp[ic * 27 + zz * 9 + yy * 3 + xx];
    }
    float s = g[oc] * rsqrtf(v[oc] + 1e-5f);
    float y = (acc + b[oc]) * s + (beta[oc] - m[oc] * s);
    x2[oc * NVOX + p0 + pp] = fmaxf(y, 0.f);
}

// ----------------------------------------------- deconv3 (x2 upsample, 140->16)
__device__ __forceinline__ int tap_list(int o, int idx[2], int kk[2]) {
    if ((o & 1) == 0) { idx[0] = o >> 1; kk[0] = 1; return 1; }
    if (o == 127)     { idx[0] = 63;     kk[0] = 0; return 1; }
    idx[0] = (o - 1) >> 1; kk[0] = 0;
    idx[1] = (o + 1) >> 1; kk[1] = 2;
    return 2;
}

__global__ __launch_bounds__(256) void k_deconv3(
    const float* __restrict__ grid, const float* __restrict__ x1,
    const float* __restrict__ x2, const float* __restrict__ cw,
    const float* __restrict__ cb, float* __restrict__ vols)
{
    int vox = blockIdx.x * 256 + threadIdx.x;
    int od = vox >> 14, oh = (vox >> 7) & 127, ow = vox & 127;
    int idz[2], kdz[2], idy[2], kdy[2], idx_[2], kdx[2];
    int nz = tap_list(od, idz, kdz);
    int ny = tap_list(oh, idy, kdy);
    int nx = tap_list(ow, idx_, kdx);
    float acc[16];
    #pragma unroll
    for (int c = 0; c < 16; ++c) acc[c] = cb[c];
    for (int a = 0; a < nz; ++a)
        for (int bb = 0; bb < ny; ++bb)
            for (int cc = 0; cc < nx; ++cc) {
                int iidx = (idz[a] * 64 + idy[bb]) * 64 + idx_[cc];
                // wt[oc][ic][k] = c3_w[ic][oc][2-k...] (flip + transpose)
                int koff = (2 - kdz[a]) * 9 + (2 - kdy[bb]) * 3 + (2 - kdx[cc]);
                for (int ic = 0; ic < 12; ++ic) {
                    float xv = grid[ic * NVOX + iidx];
                    const float* wr = cw + ic * 432 + koff;     // 432 = 16*27
                    #pragma unroll
                    for (int c = 0; c < 16; ++c) acc[c] += xv * wr[c * 27];
                }
                for (int ic = 0; ic < 64; ++ic) {
                    float xv = x1[ic * NVOX + iidx];
                    const float* wr = cw + (ic + 12) * 432 + koff;
                    #pragma unroll
                    for (int c = 0; c < 16; ++c) acc[c] += xv * wr[c * 27];
                }
                for (int ic = 0; ic < 64; ++ic) {
                    float xv = x2[ic * NVOX + iidx];
                    const float* wr = cw + (ic + 76) * 432 + koff;
                    #pragma unroll
                    for (int c = 0; c < 16; ++c) acc[c] += xv * wr[c * 27];
                }
            }
    float* o = vols + (size_t)vox * 20;          // channel-last, padded to 20
    #pragma unroll
    for (int c = 0; c < 16; ++c) o[c] = acc[c];
}

// ---------------------------------------------------------------- sobel 5x5x5
__global__ __launch_bounds__(256) void k_sobel(
    const float* __restrict__ sob, float* __restrict__ vols)
{
    int vox = blockIdx.x * 256 + threadIdx.x;
    int od = vox >> 14, oh = (vox >> 7) & 127, ow = vox & 127;
    float g0 = 0.f, g1 = 0.f, g2 = 0.f;
    for (int kd = 0; kd < 5; ++kd) {
        int qd = od + kd - 2; if ((unsigned)qd >= 128u) continue;
        for (int kh = 0; kh < 5; ++kh) {
            int qh = oh + kh - 2; if ((unsigned)qh >= 128u) continue;
            #pragma unroll
            for (int kw = 0; kw < 5; ++kw) {
                int qw = ow + kw - 2; if ((unsigned)qw >= 128u) continue;
                float dv = vols[(size_t)((qd * 128 + qh) * 128 + qw) * 20];
                int ki = (kd * 5 + kh) * 5 + kw;
                g0 += dv * sob[ki];
                g1 += dv * sob[125 + ki];
                g2 += dv * sob[250 + ki];
            }
        }
    }
    float* o = vols + (size_t)vox * 20;
    o[16] = g0; o[17] = g1; o[18] = g2;
}

// ---------------------------------------------------------------- ray setup
__global__ void k_raysetup(const float* __restrict__ ro,
                           const float* __restrict__ rd,
                           float4* __restrict__ rp)
{
    int r = blockIdx.x * 256 + threadIdx.x;
    if (r >= NRAYS) return;
    float tmn = -1e30f, tmx = 1e30f, nd = 0.f;
    #pragma unroll
    for (int c = 0; c < 3; ++c) {
        float o = ro[r * 3 + c], d = rd[r * 3 + c];
        nd += d * d;
        float vec = (d == 0.f) ? 1e-6f : d;
        float ra = (1.f - o) / vec, rb = (-1.f - o) / vec;
        tmn = fmaxf(tmn, fminf(ra, rb));
        tmx = fminf(tmx, fmaxf(ra, rb));
    }
    float tmin = fminf(fmaxf(tmn, 0.05f), 2.5f);
    float tmax = fminf(fmaxf(tmx, 0.05f), 2.5f);
    rp[r] = make_float4(tmin, sqrtf(nd), (tmax <= tmin) ? 1.f : 0.f, 0.f);
}

// ------------------------------------ pack MLP weights, TRANSPOSED, to f16
// Wt[n_out][K] row-major so WMMA A-fragments (W^T tiles) are LDS-contiguous.
__global__ void k_pack(const float* __restrict__ w0, const float* __restrict__ w1,
                       const float* __restrict__ w2, const float* __restrict__ w3,
                       h16* __restrict__ W0, h16* __restrict__ W1,
                       h16* __restrict__ W2, h16* __restrict__ W3)
{
    int i = blockIdx.x * 256 + threadIdx.x;
    if (i < 4096) {
        int n = i >> 6, k = i & 63;            // Wt[n*64 + k] = w[k][n]
        W0[i] = (h16)((k < 58) ? w0[k * 64 + n] : 0.f);  // pad K 58->64
        W1[i] = (h16)w1[k * 64 + n];
        W2[i] = (h16)w2[k * 64 + n];
    }
    if (i < 1024) {
        int n = i >> 6, k = i & 63;            // 16 rows, pad N 3->16 with zeros
        W3[i] = (h16)((n < 3) ? w3[k * 3 + n] : 0.f);
    }
}

// ---------------- contiguous 16-half LDS load -> v16h fragment helpers -------
__device__ __forceinline__ v16h ld16(const h16* p) {          // 16 contiguous
    v8h lo = *(const v8h*)p;
    v8h hh = *(const v8h*)(p + 8);
    return __builtin_shufflevector(lo, hh, 0,1,2,3,4,5,6,7,8,9,10,11,12,13,14,15);
}
__device__ __forceinline__ v16h ld_split16(const h16* p) {    // two 8-half chunks
    v8h lo = *(const v8h*)p;                                  // (A-matrix layout)
    v8h hh = *(const v8h*)(p + 16);
    return __builtin_shufflevector(lo, hh, 0,1,2,3,4,5,6,7,8,9,10,11,12,13,14,15);
}

// --------------------- one WMMA MLP layer:  H^T = W^T (A)  x  X^T (B) --------
// act: [sample 0..31][K 0..63] f16 row-major, Wt: [n_out][64] f16 row-major.
// A frag (16x32): lane(m,hi) elem E -> K=E+16hi                  (2x b128 loads)
// B frag (32x16): elem E -> K=E+16hi, col=sample                (2x b128 loads)
// D: lane holds rows n_out=j+8hi, col=sample -> b128 store per tile.
__device__ __forceinline__ void mlp_layer(
    const h16* __restrict__ act, const h16* __restrict__ Wt,
    const float* __restrict__ bias, int mtiles, bool relu,
    h16* __restrict__ actOut, float* __restrict__ rgbOut)
{
    int lane = threadIdx.x & 31;
    int m = lane & 15, hi = lane >> 4;
    v16h b[2][2];
    #pragma unroll
    for (int nt = 0; nt < 2; ++nt)
        #pragma unroll
        for (int kc = 0; kc < 2; ++kc)
            b[nt][kc] = ld16(act + (nt * 16 + m) * 64 + kc * 32 + 16 * hi);
    #pragma unroll
    for (int mt = 0; mt < mtiles; ++mt) {
        v16h a0 = ld16(Wt + (mt * 16 + m) * 64 + 16 * hi);
        v16h a1 = ld16(Wt + (mt * 16 + m) * 64 + 32 + 16 * hi);
        #pragma unroll
        for (int nt = 0; nt < 2; ++nt) {
            v8f c;
            #pragma unroll
            for (int j = 0; j < 8; ++j) c[j] = bias[mt * 16 + 8 * hi + j];
            c = __builtin_amdgcn_wmma_f32_16x16x32_f16(false, a0, false, b[nt][0],
                                                       (short)0, c, false, false);
            c = __builtin_amdgcn_wmma_f32_16x16x32_f16(false, a1, false, b[nt][1],
                                                       (short)0, c, false, false);
            if (rgbOut) {                 // rgb = rows 0..2 -> hi==0, j<3
                if (hi == 0) {
                    #pragma unroll
                    for (int j = 0; j < 3; ++j)
                        rgbOut[(nt * 16 + m) * 4 + j] = c[j];
                }
            } else {
                v8h o;
                #pragma unroll
                for (int j = 0; j < 8; ++j) {
                    float vv = c[j];
                    if (relu) vv = fmaxf(vv, 0.f);
                    o[j] = (h16)vv;
                }
                *(v8h*)(actOut + (nt * 16 + m) * 64 + mt * 16 + 8 * hi) = o;
            }
        }
    }
}

// -------------------- fused: trilinear sample + embedding + WMMA MLP ---------
__global__ __launch_bounds__(128) void k_sample_mlp(
    const float* __restrict__ ro, const float* __restrict__ rd,
    const float* __restrict__ vd, const float* __restrict__ vols,
    const float4* __restrict__ rp,
    const h16* __restrict__ pW0, const h16* __restrict__ pW1,
    const h16* __restrict__ pW2, const h16* __restrict__ pW3,
    const float* __restrict__ b0, const float* __restrict__ b1,
    const float* __restrict__ b2, const float* __restrict__ b3,
    float* __restrict__ sdf_out, float4* __restrict__ rgb_out,
    float* __restrict__ angle_out)
{
    __shared__ alignas(16) h16  sW0[4096], sW1[4096], sW2[4096], sW3[1024];
    __shared__ float sB0[64], sB1[64], sB2[64], sB3[16];
    __shared__ alignas(16) h16  sActA[4][2048];   // 4 waves x [32 samples][64]
    __shared__ alignas(16) h16  sActB[4][2048];
    __shared__ float sRGB[4][128];                // [32 samples][4]

    int tid = threadIdx.x;
    {   // vectorized weight staging (uint4 = 8 halves)
        const uint4* g0 = (const uint4*)pW0; const uint4* g1 = (const uint4*)pW1;
        const uint4* g2 = (const uint4*)pW2; const uint4* g3 = (const uint4*)pW3;
        uint4* s0 = (uint4*)sW0; uint4* s1 = (uint4*)sW1;
        uint4* s2 = (uint4*)sW2; uint4* s3 = (uint4*)sW3;
        for (int i = tid; i < 512; i += 128) { s0[i] = g0[i]; s1[i] = g1[i]; s2[i] = g2[i]; }
        if (tid < 128) s3[tid] = g3[tid];
    }
    if (tid < 64) { sB0[tid] = b0[tid]; sB1[tid] = b1[tid]; sB2[tid] = b2[tid]; }
    if (tid < 16) sB3[tid] = (tid < 3) ? b3[tid] : 0.f;
    __syncthreads();

    int s    = blockIdx.x * 128 + tid;          // grid sized exactly: no guard
    int ray  = s / NSAMP;
    int samp = s - ray * NSAMP;
    int wid  = tid >> 5, lane = tid & 31;

    float ox = ro[ray * 3], oy = ro[ray * 3 + 1], oz = ro[ray * 3 + 2];
    float dx = rd[ray * 3], dy = rd[ray * 3 + 1], dz = rd[ray * 3 + 2];
    float vx = vd[ray * 3], vy = vd[ray * 3 + 1], vz = vd[ray * 3 + 2];
    float4 pr = rp[ray];
    float it = pr.x + (STEPVOX * (float)samp) / pr.y;
    float px = ox + dx * it, py = oy + dy * it, pz = oz + dz * it;
    bool mask = (pr.z != 0.f) || (px < -1.f) || (px > 1.f) ||
                (py < -1.f) || (py > 1.f) || (pz < -1.f) || (pz > 1.f);

    float fx = fminf(fmaxf((px + 1.f) * 63.5f, 0.f), 127.f);
    float fy = fminf(fmaxf((py + 1.f) * 63.5f, 0.f), 127.f);
    float fz = fminf(fmaxf((pz + 1.f) * 63.5f, 0.f), 127.f);
    int ix0 = (int)floorf(fx), iy0 = (int)floorf(fy), iz0 = (int)floorf(fz);
    float tx = fx - (float)ix0, ty = fy - (float)iy0, tz = fz - (float)iz0;
    int ix1 = min(ix0 + 1, 127), iy1 = min(iy0 + 1, 127), iz1 = min(iz0 + 1, 127);

    float a20[20];
    #pragma unroll
    for (int c = 0; c < 20; ++c) a20[c] = 0.f;
    #pragma unroll
    for (int cx = 0; cx < 2; ++cx) {
        int ixx = cx ? ix1 : ix0; float wx = cx ? tx : 1.f - tx;
        #pragma unroll
        for (int cy = 0; cy < 2; ++cy) {
            int iyy = cy ? iy1 : iy0; float wy = cy ? ty : 1.f - ty;
            #pragma unroll
            for (int cz = 0; cz < 2; ++cz) {
                int izz = cz ? iz1 : iz0; float wz = cz ? tz : 1.f - tz;
                float wgt = wx * wy * wz;
                const float4* cp = reinterpret_cast<const float4*>(
                    vols + (size_t)((ixx * 128 + iyy) * 128 + izz) * 20);
                #pragma unroll
                for (int q = 0; q < 5; ++q) {        // 80B/voxel -> b128 loads
                    float4 vvv = cp[q];
                    a20[q * 4 + 0] += vvv.x * wgt; a20[q * 4 + 1] += vvv.y * wgt;
                    a20[q * 4 + 2] += vvv.z * wgt; a20[q * 4 + 3] += vvv.w * wgt;
                }
            }
        }
    }
    sdf_out[s] = mask ? 100.f : a20[0];

    float g0 = a20[16], g1 = a20[17], g2 = a20[18];
    float gn = fmaxf(sqrtf(g0 * g0 + g1 * g1 + g2 * g2), 1e-12f);
    float nxx = mask ? 0.f : g0 / gn;
    float nyy = mask ? 0.f : g1 / gn;
    float nzz = mask ? 0.f : g2 / gn;
    float dot_vn = -(vx * nxx + vy * nyy + vz * nzz);
    if (samp == NSAMP - 1) angle_out[ray] = -dot_vn;   // angle_n_d last sample
    float rx = 2.f * dot_vn * nxx + vx;
    float ry = 2.f * dot_vn * nyy + vy;
    float rz = 2.f * dot_vn * nzz + vz;

    // build 58-dim input in registers, then 8x b128 LDS stores
    h16 tmp[64];
    #pragma unroll
    for (int c = 0; c < 15; ++c) tmp[c] = (h16)a20[1 + c];   // feat
    tmp[15] = (h16)dot_vn;
    tmp[16] = (h16)rx; tmp[17] = (h16)ry; tmp[18] = (h16)rz; // emb: refdirs
    float fr = 1.f;
    #pragma unroll
    for (int fi = 0; fi < 6; ++fi) {                         // sin/cos freqs
        tmp[19 + fi * 6 + 0] = (h16)sinf(rx * fr);
        tmp[19 + fi * 6 + 1] = (h16)sinf(ry * fr);
        tmp[19 + fi * 6 + 2] = (h16)sinf(rz * fr);
        tmp[19 + fi * 6 + 3] = (h16)cosf(rx * fr);
        tmp[19 + fi * 6 + 4] = (h16)cosf(ry * fr);
        tmp[19 + fi * 6 + 5] = (h16)cosf(rz * fr);
        fr *= 2.f;
    }
    tmp[55] = (h16)px; tmp[56] = (h16)py; tmp[57] = (h16)pz; // pts
    #pragma unroll
    for (int c = 58; c < 64; ++c) tmp[c] = (h16)0.f;         // K pad
    v8h* Av = (v8h*)(&sActA[wid][lane * 64]);
    #pragma unroll
    for (int i = 0; i < 8; ++i) {
        v8h t;
        #pragma unroll
        for (int j = 0; j < 8; ++j) t[j] = tmp[i * 8 + j];
        Av[i] = t;
    }

    __syncthreads();
    mlp_layer(sActA[wid], sW0, sB0, 4, true,  sActB[wid], nullptr);
    __syncthreads();
    mlp_layer(sActB[wid], sW1, sB1, 4, true,  sActA[wid], nullptr);
    __syncthreads();
    mlp_layer(sActA[wid], sW2, sB2, 4, true,  sActB[wid], nullptr);
    __syncthreads();
    mlp_layer(sActB[wid], sW3, sB3, 1, false, nullptr,   sRGB[wid]);
    __syncthreads();

    float r0 = sRGB[wid][lane * 4 + 0];
    float r1 = sRGB[wid][lane * 4 + 1];
    float r2 = sRGB[wid][lane * 4 + 2];
    if (mask) { r0 = 0.f; r1 = 0.f; r2 = 0.f; }
    rgb_out[s] = make_float4(r0, r1, r2, 0.f);
}

// ---------------------------------------------------------------- compositing
__global__ void k_composite(const float* __restrict__ sdf,
                            const float4* __restrict__ rgbs,
                            const float4* __restrict__ rp,
                            const float* __restrict__ variance,
                            float* __restrict__ out)
{
    int r = blockIdx.x * 256 + threadIdx.x;
    if (r >= NRAYS) return;
    float scale = expf(variance[0] * 10.f);
    float4 pr = rp[r];
    float dbase = pr.x * pr.y;                    // t_min * ||d||
    float sd_prev = 1.f / (1.f + expf(-sdf[r * NSAMP] * scale));
    float T = 1.f, cr = 0.f, cg = 0.f, cb = 0.f, depth = 0.f;
    float* wout = out + 24576 + (size_t)r * (NSAMP - 1);
    for (int i = 0; i < NSAMP - 1; ++i) {
        float sd_next = 1.f / (1.f + expf(-sdf[r * NSAMP + i + 1] * scale));
        float alpha = fmaxf((sd_prev - sd_next) / (sd_prev + 1e-10f), 0.f);
        float w = T * alpha;
        wout[i] = w;
        float4 c = rgbs[r * NSAMP + i];
        cr += w * c.x; cg += w * c.y; cb += w * c.z;
        depth += w * (dbase + STEPVOX * (float)i);
        T *= (1.f - alpha);
        sd_prev = sd_next;
    }
    out[r * 3 + 0] = cr; out[r * 3 + 1] = cg; out[r * 3 + 2] = cb;
    out[12288 + r] = depth;
    out[16384 + r] = T;                           // bg = T[..., -1]
}

// ---------------------------------------------------------------------- launch
extern "C" void kernel_launch(void* const* d_in, const int* in_sizes, int n_in,
                              void* d_out, int out_size, void* d_ws, size_t ws_size,
                              hipStream_t stream)
{
    const float* rays_o   = (const float*)d_in[0];
    const float* rays_d   = (const float*)d_in[1];
    const float* viewdirs = (const float*)d_in[2];
    const float* grid     = (const float*)d_in[3];
    const float* c1_w = (const float*)d_in[4];   const float* c1_b = (const float*)d_in[5];
    const float* bn1_g = (const float*)d_in[6];  const float* bn1_beta = (const float*)d_in[7];
    const float* bn1_m = (const float*)d_in[8];  const float* bn1_v = (const float*)d_in[9];
    const float* c2_w = (const float*)d_in[10];  const float* c2_b = (const float*)d_in[11];
    const float* bn2_g = (const float*)d_in[12]; const float* bn2_beta = (const float*)d_in[13];
    const float* bn2_m = (const float*)d_in[14]; const float* bn2_v = (const float*)d_in[15];
    const float* c3_w = (const float*)d_in[16];  const float* c3_b = (const float*)d_in[17];
    const float* sobel = (const float*)d_in[18]; const float* variance = (const float*)d_in[19];
    const float* w0 = (const float*)d_in[20];    const float* b0 = (const float*)d_in[21];
    const float* w1 = (const float*)d_in[22];    const float* b1 = (const float*)d_in[23];
    const float* w2 = (const float*)d_in[24];    const float* b2 = (const float*)d_in[25];
    const float* w3 = (const float*)d_in[26];    const float* b3 = (const float*)d_in[27];
    (void)in_sizes; (void)n_in; (void)out_size; (void)ws_size;

    char* ws = (char*)d_ws;
    float*  x1      = (float*)(ws);                            // 64ch * 64^3 (64MB)
    float*  x2      = (float*)(ws + (size_t)67108864);         // 64ch * 64^3 (64MB)
    float*  vols    = (float*)(ws + (size_t)134217728);        // 128^3 * 20 f32 (160MB)
    float*  sdf_buf = (float*)(ws);                            // alias x1 (dead)
    float4* rgb_buf = (float4*)(ws + (size_t)8388608);         // alias x1 + 8MB
    float4* rp      = (float4*)(ws + (size_t)67108864);        // alias x2 (dead)
    h16* pW0 = (h16*)(ws + (size_t)301989888);
    h16* pW1 = pW0 + 4096;
    h16* pW2 = pW1 + 4096;
    h16* pW3 = pW2 + 4096;

    float* out = (float*)d_out;

    k_pack    <<<16,    256, 0, stream>>>(w0, w1, w2, w3, pW0, pW1, pW2, pW3);
    k_conv1   <<<65536, 256, 0, stream>>>(grid, c1_w, c1_b, bn1_g, bn1_beta, bn1_m, bn1_v, x1);
    k_conv2   <<<65536, 256, 0, stream>>>(grid, x1, c2_w, c2_b, bn2_g, bn2_beta, bn2_m, bn2_v, x2);
    k_deconv3 <<<8192,  256, 0, stream>>>(grid, x1, x2, c3_w, c3_b, vols);
    k_sobel   <<<8192,  256, 0, stream>>>(sobel, vols);
    k_raysetup<<<16,    256, 0, stream>>>(rays_o, rays_d, rp);
    k_sample_mlp<<<14304, 128, 0, stream>>>(rays_o, rays_d, viewdirs, vols, rp,
                                            pW0, pW1, pW2, pW3, b0, b1, b2, b3,
                                            sdf_buf, rgb_buf, out + 20480);
    k_composite<<<16,   256, 0, stream>>>(sdf_buf, rgb_buf, rp, variance, out);
}
